// Teacher_Model_41807211659637
// MI455X (gfx1250) — matrix operand order repaired and checked by reading, not compiled, not run on hardware
//
#include <hip/hip_runtime.h>
#include <hip/hip_bf16.h>
#include <math.h>

// CDNA5 / gfx1250 implementation of the GIN + persistence-image pipeline.
// Dense 32x32 / 64x32 GEMMs use v_wmma_f32_16x16x32_f16 (f32 accumulate).
// Weights are pre-packed into WMMA fragment layout so each lane loads its
// whole B fragment with contiguous b128 vector loads.

typedef __attribute__((ext_vector_type(16))) _Float16 v16h;
typedef __attribute__((ext_vector_type(8)))  float    v8f;

#define HID 32
#define WPB 8          // waves per block for WMMA kernels
#define RES 5
#define PRELU_A 0.1f
#define LPAD 33        // LDS row stride (floats) to avoid bank conflicts

// ---- CDNA5 WMMA fragment index helpers (wave32, 16x16x32 f16) ----
// A (16x32, f16): lanes 0-15 hold row M=lane, K in {0..7,16..23};
//                 lanes 16-31 hold row M=lane-16, K in {8..15,24..31}.
__device__ __forceinline__ int a_kmap(int e, int lane) {
  return ((e < 8) ? e : (e + 8)) + ((lane >> 4) << 3);
}
// B (32x16, f16): lanes 0-15 hold col N=lane, K in {0..7,16..23};
//                 lanes 16-31 hold col N=lane-16, K in {8..15,24..31}.
__device__ __forceinline__ int b_kmap(int e, int lane) {
  return ((e < 8) ? e : (e + 8)) + ((lane >> 4) << 3);
}

// Load one 32-float row into an A fragment (b128 loads, cvt to f16).
__device__ __forceinline__ v16h load_a_row(const float* __restrict__ row, int lane) {
  const float4* rv = (const float4*)row;
  int q = (lane >> 4) * 2;            // kb/4 : 0 or 2
  float4 f0 = rv[q];                  // K = kb .. kb+3
  float4 f1 = rv[q + 1];              // K = kb+4 .. kb+7
  float4 f2 = rv[q + 4];              // K = 16+kb .. 16+kb+3
  float4 f3 = rv[q + 5];              // K = 16+kb+4 .. 16+kb+7
  v16h a;
  a[0]  = (_Float16)f0.x; a[1]  = (_Float16)f0.y; a[2]  = (_Float16)f0.z; a[3]  = (_Float16)f0.w;
  a[4]  = (_Float16)f1.x; a[5]  = (_Float16)f1.y; a[6]  = (_Float16)f1.z; a[7]  = (_Float16)f1.w;
  a[8]  = (_Float16)f2.x; a[9]  = (_Float16)f2.y; a[10] = (_Float16)f2.z; a[11] = (_Float16)f2.w;
  a[12] = (_Float16)f3.x; a[13] = (_Float16)f3.y; a[14] = (_Float16)f3.z; a[15] = (_Float16)f3.w;
  return a;
}

// ---------------- weight pre-packing into fragment layout ----------------
// Packed layout: P[((nt*32 + lane)*16) + e] = (f16) W[b_kmap(e,lane)*32 + nt*16 + (lane&15)]
__global__ void k_pack32(const float* __restrict__ W, _Float16* __restrict__ P) {
  int t = threadIdx.x + blockIdx.x * blockDim.x;       // 1024 elements
  if (t < 1024) {
    int e = t & 15;
    int lane = (t >> 4) & 31;
    int nt = t >> 9;
    P[t] = (_Float16)W[b_kmap(e, lane) * HID + nt * 16 + (lane & 15)];
  }
}
// W5 is 64x32: half 0 packs rows 0..31, half 1 packs rows 32..63 (2048 elements).
__global__ void k_pack64(const float* __restrict__ W5, _Float16* __restrict__ P) {
  int t = threadIdx.x + blockIdx.x * blockDim.x;       // 2048 elements
  if (t < 2048) {
    int half = t >> 10;
    int rem = t & 1023;
    int e = rem & 15;
    int lane = (rem >> 4) & 31;
    int nt = rem >> 9;
    P[t] = (_Float16)W5[(b_kmap(e, lane) + half * 32) * HID + nt * 16 + (lane & 15)];
  }
}

// ---------------- utility kernels ----------------
__global__ void k_copy(float* __restrict__ dst, const float* __restrict__ src, long long n) {
  long long i = (long long)blockIdx.x * blockDim.x + threadIdx.x;
  long long stride = (long long)gridDim.x * blockDim.x;
  for (; i < n; i += stride) dst[i] = src[i];
}

// agg[dst[i]] += x[src[i]] over all EN edges (scalar features, layer 1)
__global__ void k_scatter1(const float* __restrict__ x, const int* __restrict__ src,
                           const int* __restrict__ dst, float* __restrict__ agg, int EN) {
  int i = blockIdx.x * blockDim.x + threadIdx.x;
  if (i < EN) atomicAdd(&agg[dst[i]], x[src[i]]);
}

// h1[n][j] = relu(agg[n]*W1[j] + b1[j])
__global__ void k_layer1(const float* __restrict__ agg, const float* __restrict__ W1,
                         const float* __restrict__ b1, float* __restrict__ Y, int Nn) {
  int t = blockIdx.x * blockDim.x + threadIdx.x;
  if (t < Nn * HID) {
    int n = t >> 5, j = t & 31;
    float v = fmaf(agg[n], W1[j], b1[j]);
    Y[t] = v > 0.f ? v : 0.f;
  }
}

// Y[dst[i]*32+j] += X[src[i]*32+j]; one wave per edge, lane = feature j.
// Edge index is wave-uniform -> keep it scalar via readfirstlane.
__global__ void k_scatter32(const float* __restrict__ X, const int* __restrict__ src,
                            const int* __restrict__ dst, float* __restrict__ Y, int EN) {
  int i = blockIdx.x * (blockDim.x >> 5) + (threadIdx.x >> 5);
  int j = threadIdx.x & 31;
  if (i < EN) {
    int s = __builtin_amdgcn_readfirstlane(src[i]);
    int d = __builtin_amdgcn_readfirstlane(dst[i]);
    atomicAdd(&Y[(long long)d * HID + j], X[(long long)s * HID + j]);
  }
}

// ---------------- node GEMM: Y = act(X @ W + b), X:(Nn,32) W packed ----------------
template <int RELU>
__global__ void k_gemm32(const float* __restrict__ X, const _Float16* __restrict__ PW,
                         const float* __restrict__ bias, float* __restrict__ Y,
                         int Nn) {
  int lane = threadIdx.x & 31;
  int wave = threadIdx.x >> 5;
  int tiles = (Nn + 15) >> 4;
  int tile = blockIdx.x * WPB + wave;
  if (tile >= tiles) return;                   // wave-uniform: EXEC stays all-1s inside

  int m0 = tile * 16;
  int ma = m0 + (lane & 15);
  if (ma >= Nn) ma = Nn - 1;                   // clamp tail loads (Nn%16==0 normally)
  v16h a = load_a_row(X + (long long)ma * HID, lane);

  int nloc = lane & 15;
  int moff = (lane >> 4) << 3;
  bool full = (m0 + 16 <= Nn);                 // wave-uniform
#pragma unroll
  for (int nt = 0; nt < 2; ++nt) {
    int ncol = nt * 16 + nloc;
    v16h b = *(const v16h*)(PW + ((nt * 32 + lane) << 4));
    v8f c;
    float bv = bias[ncol];
#pragma unroll
    for (int r = 0; r < 8; ++r) c[r] = bv;
    c = __builtin_amdgcn_wmma_f32_16x16x32_f16(false, a, false, b, (short)0, c, false, false);
    float* __restrict__ yp = Y + (long long)(m0 + moff) * HID + ncol;
    if (full) {
#pragma unroll
      for (int r = 0; r < 8; ++r) {
        float v = c[r];
        if (RELU) v = fmaxf(v, 0.f);
        yp[r * HID] = v;
      }
    } else {
#pragma unroll
      for (int r = 0; r < 8; ++r) {
        float v = c[r];
        if (RELU) v = fmaxf(v, 0.f);
        if (m0 + moff + r < Nn) yp[r * HID] = v;
      }
    }
  }
}

// ------------- edge MLP: h = prelu([x_src,x_dst]@W5+b5); pd = h@W6+b6 -------------
__global__ void k_edge(const float* __restrict__ X, const int* __restrict__ src,
                       const int* __restrict__ dst,
                       const _Float16* __restrict__ PW5, const float* __restrict__ b5,
                       const float* __restrict__ W6, const float* __restrict__ b6,
                       float* __restrict__ pd, int E) {
  __shared__ float hl[WPB][16 * LPAD];
  int lane = threadIdx.x & 31;
  int wave = threadIdx.x >> 5;
  int tiles = (E + 15) >> 4;
  int tile = blockIdx.x * WPB + wave;
  bool active = (tile < tiles);
  int e0 = active ? tile * 16 : 0;

  if (active) {
    int ml = lane & 15;
    int eid = e0 + ml;
    if (eid >= E) eid = E - 1;
    int s = src[eid], d = dst[eid];
    v16h a_in  = load_a_row(X + (long long)s * HID, lane);   // K 0..31  (x_in)
    v16h a_out = load_a_row(X + (long long)d * HID, lane);   // K 32..63 (x_out)
    int nloc = lane & 15;
    int moff = (lane >> 4) << 3;
#pragma unroll
    for (int nt = 0; nt < 2; ++nt) {
      int ncol = nt * 16 + nloc;
      v16h b1 = *(const v16h*)(PW5 + ((nt * 32 + lane) << 4));          // rows 0..31
      v16h b2 = *(const v16h*)(PW5 + 1024 + ((nt * 32 + lane) << 4));   // rows 32..63
      v8f c;
      float bv = b5[ncol];
#pragma unroll
      for (int r = 0; r < 8; ++r) c[r] = bv;
      c = __builtin_amdgcn_wmma_f32_16x16x32_f16(false, a_in,  false, b1, (short)0, c, false, false);
      c = __builtin_amdgcn_wmma_f32_16x16x32_f16(false, a_out, false, b2, (short)0, c, false, false);
#pragma unroll
      for (int r = 0; r < 8; ++r) {
        float v = c[r];
        v = v >= 0.f ? v : PRELU_A * v;                    // PReLU
        hl[wave][(moff + r) * LPAD + ncol] = v;
      }
    }
  }
  __syncthreads();
  if (active && lane < 16) {
    int eid = e0 + lane;
    if (eid < E) {
      float p0 = b6[0], p1 = b6[1];
#pragma unroll
      for (int n = 0; n < HID; ++n) {
        float h = hl[wave][lane * LPAD + n];
        p0 = fmaf(h, W6[n * 2 + 0], p0);
        p1 = fmaf(h, W6[n * 2 + 1], p1);
      }
      pd[(long long)eid * 2 + 0] = p0;
      pd[(long long)eid * 2 + 1] = p1;
    }
  }
}

// ---------------- persistence image ----------------
__global__ void k_img_zero(float* img) {
  if (threadIdx.x < RES * RES) img[threadIdx.x] = 0.f;
}

__global__ void k_image(const float* __restrict__ pd, float* __restrict__ img, int E) {
  __shared__ float simg[RES * RES];
  int t = threadIdx.x;
  if (t < RES * RES) simg[t] = 0.f;
  __syncthreads();

  float acc[RES * RES];
#pragma unroll
  for (int k = 0; k < RES * RES; ++k) acc[k] = 0.f;

  const float inv = 3.5355339059f;                 // 1/(0.2*sqrt(2))
  for (long long e = (long long)blockIdx.x * blockDim.x + t; e < E;
       e += (long long)gridDim.x * blockDim.x) {
    float b = pd[2 * e];
    float pers = pd[2 * e + 1] - b;
    float cb[RES + 1], cp[RES + 1];
#pragma unroll
    for (int k = 0; k <= RES; ++k) {
      float ek = 0.2f * (float)k;
      cb[k] = 0.5f * (1.f + erff((ek - b) * inv));
      cp[k] = 0.5f * (1.f + erff((ek - pers) * inv));
    }
#pragma unroll
    for (int i = 0; i < RES; ++i) {
      float dbi = pers * (cb[i + 1] - cb[i]);
#pragma unroll
      for (int j = 0; j < RES; ++j)
        acc[i * RES + j] = fmaf(dbi, cp[j + 1] - cp[j], acc[i * RES + j]);
    }
  }
#pragma unroll
  for (int k = 0; k < RES * RES; ++k) atomicAdd(&simg[k], acc[k]);
  __syncthreads();
  if (t < RES * RES) atomicAdd(&img[t], simg[t]);
}

// ---------------- launch ----------------
extern "C" void kernel_launch(void* const* d_in, const int* in_sizes, int n_in,
                              void* d_out, int out_size, void* d_ws, size_t ws_size,
                              hipStream_t stream) {
  const float* x0  = (const float*)d_in[0];
  const int*   ei  = (const int*)d_in[1];
  const float* W1  = (const float*)d_in[2];
  const float* b1  = (const float*)d_in[3];
  const float* W2  = (const float*)d_in[4];
  const float* b2  = (const float*)d_in[5];
  const float* W4  = (const float*)d_in[6];
  const float* b4  = (const float*)d_in[7];
  const float* W3  = (const float*)d_in[8];
  const float* b3  = (const float*)d_in[9];
  const float* W5  = (const float*)d_in[10];
  const float* b5  = (const float*)d_in[11];
  const float* W6  = (const float*)d_in[12];
  const float* b6  = (const float*)d_in[13];

  int Nn = in_sizes[0];            // 100000
  int EN = in_sizes[1] / 2;        // E + N (self-loops appended)
  int E  = EN - Nn;                // 1600000
  const int* src = ei;
  const int* dstI = ei + EN;

  float* pd  = (float*)d_out;                        // (E,2) row-major
  float* img = (float*)d_out + (out_size - RES * RES);

  // workspace: scalar agg (padded) + two (Nn x 32) ping-pong buffers + packed weights
  float* aggS = (float*)d_ws;
  long long npad = ((long long)Nn + 255) & ~255LL;
  float* bufA = aggS + npad;
  float* bufB = bufA + (long long)Nn * HID;
  _Float16* pack = (_Float16*)(bufB + (long long)Nn * HID);
  _Float16* pW2 = pack;              // 1024
  _Float16* pW4 = pW2 + 1024;        // 1024
  _Float16* pW3 = pW4 + 1024;        // 1024
  _Float16* pW5 = pW3 + 1024;        // 2048

  // ---- pack weights into WMMA fragment layout (tiny, once per launch) ----
  k_pack32<<<4, 256, 0, stream>>>(W2, pW2);
  k_pack32<<<4, 256, 0, stream>>>(W4, pW4);
  k_pack32<<<4, 256, 0, stream>>>(W3, pW3);
  k_pack64<<<8, 256, 0, stream>>>(W5, pW5);

  // ---- layer 1: agg = x0 + segsum(x0[src], dst); h1 = relu(agg*W1 + b1) ----
  k_copy<<<2048, 256, 0, stream>>>(aggS, x0, Nn);
  k_scatter1<<<(EN + 255) / 256, 256, 0, stream>>>(x0, src, dstI, aggS, EN);
  k_layer1<<<(Nn * HID + 255) / 256, 256, 0, stream>>>(aggS, W1, b1, bufA, Nn);

  // ---- layers 2..4: agg32 then WMMA GEMM (relu, relu, linear) ----
  const _Float16* Ps[3] = {pW2, pW4, pW3};
  const float*    bs[3] = {b2, b4, b3};
  long long scatterThreads = (long long)EN * HID;
  int scatterBlocks = (int)((scatterThreads + 255) / 256);
  int tilesN = (Nn + 15) >> 4;
  int gemmBlocks = (tilesN + WPB - 1) / WPB;
  for (int l = 0; l < 3; ++l) {
    k_copy<<<4096, 256, 0, stream>>>(bufB, bufA, (long long)Nn * HID);
    k_scatter32<<<scatterBlocks, 256, 0, stream>>>(bufA, src, dstI, bufB, EN);
    if (l < 2)
      k_gemm32<1><<<gemmBlocks, 32 * WPB, 0, stream>>>(bufB, Ps[l], bs[l], bufA, Nn);
    else
      k_gemm32<0><<<gemmBlocks, 32 * WPB, 0, stream>>>(bufB, Ps[l], bs[l], bufA, Nn);
  }

  // ---- edge MLP -> pd ----
  int tilesE = (E + 15) >> 4;
  int edgeBlocks = (tilesE + WPB - 1) / WPB;
  k_edge<<<edgeBlocks, 32 * WPB, 0, stream>>>(bufA, src, dstI, pW5, b5, W6, b6, pd, E);

  // ---- persistence image -> img ----
  k_img_zero<<<1, 32, 0, stream>>>(img);
  k_image<<<2048, 256, 0, stream>>>(pd, img, E);
}
